// Mamba2_3745211482481
// MI455X (gfx1250) — matrix operand order repaired and checked by reading, not compiled, not run on hardware
//
#include <hip/hip_runtime.h>
#include <hip/hip_bf16.h>
#include <stdint.h>

// ---------------- CDNA5 WMMA types ----------------
typedef __attribute__((ext_vector_type(16))) __bf16 v16bf;
typedef __attribute__((ext_vector_type(8)))  float  v8f;

#define D_MODEL   2048
#define D_STATE   128
#define HEADDIM   64
#define NHEADS    64
#define D_INNER   4096
#define CONV_DIM  4352
#define D_IN_PROJ 8512
#define SEQLEN    4096
#define CHUNK     256
#define NCHUNK    16
#define EPS       1e-5f

// CDNA5 async global->LDS path (ASYNCcnt), guarded so either toolchain compiles.
// Builtin signature (from clang diagnostic): arg0 = int4(vector_size) pointer in
// global AS (non-const), arg1 = int4 pointer in LDS AS, then (imm offset, cpol).
#if defined(__HIP_DEVICE_COMPILE__) && \
    __has_builtin(__builtin_amdgcn_global_load_async_to_lds_b128) && \
    __has_builtin(__builtin_amdgcn_s_wait_asynccnt)
#define HAVE_ASYNC_LDS 1
typedef int v4i_vs __attribute__((vector_size(16)));
typedef v4i_vs __attribute__((address_space(1)))* g_v4i_p;
typedef v4i_vs __attribute__((address_space(3)))* l_v4i_p;
#else
#define HAVE_ASYNC_LDS 0
#endif

// ---------------- helpers ----------------
__device__ __forceinline__ unsigned short f2bf(float f) {
    unsigned u = __float_as_uint(f);
    unsigned r = u + 0x7FFFu + ((u >> 16) & 1u);   // round-to-nearest-even
    return (unsigned short)(r >> 16);
}
__device__ __forceinline__ float bf2f(unsigned short b) {
    return __uint_as_float(((unsigned)b) << 16);
}
__device__ __forceinline__ float silu(float x) { return x / (1.0f + __expf(-x)); }
__device__ __forceinline__ float softplus_f(float x) {
    return (x > 20.0f) ? x : log1pf(expf(x));
}

union FragU { v16bf v; uint4 q[2]; };

// A-operand (16x32 bf16): lane L<16 holds row M=L, K = {koff..koff+7, koff+16..koff+23}
// with koff = (lane>>4)*8.  base points at (row0, k0), ld in elements (mult of 8).
__device__ __forceinline__ v16bf load_frag_a(const unsigned short* base, int ld, int lane) {
    int r = lane & 15, half = lane >> 4;
    const unsigned short* p = base + (size_t)r * ld + half * 8;
    FragU f;
    f.q[0] = *(const uint4*)(p);
    f.q[1] = *(const uint4*)(p + 16);
    return f.v;
}
// B-operand (32x16 bf16): lane holds column N=lane&15, K = {koff..koff+7, koff+16..koff+23}, koff = (lane>>4)*8.
// base points at (col0, k0) of a [col][k] row-major array (ld elements, mult of 8).
__device__ __forceinline__ v16bf load_frag_b(const unsigned short* base, int ld, int lane) {
    int c = lane & 15, half = lane >> 4;
    const unsigned short* p = base + (size_t)c * ld + half * 8;
    FragU f;
    f.q[0] = *(const uint4*)(p);
    f.q[1] = *(const uint4*)(p + 16);
    return f.v;
}
__device__ __forceinline__ v8f wmma_bf16(v16bf a, v16bf b, v8f c) {
    return __builtin_amdgcn_wmma_f32_16x16x32_bf16(false, a, false, b, (short)0, c, false, false);
}

// ---------------- kernel 0: f32 -> bf16 cast ----------------
__global__ void k_cast_bf16(const float* __restrict__ src, unsigned short* __restrict__ dst, int n) {
    int i = blockIdx.x * 256 + threadIdx.x;
    if (i < n) dst[i] = f2bf(src[i]);
}

// ---------------- kernel 1: generic WMMA GEMM  C[M,N] = A[M,K] * B[N,K]^T ----------------
// M % 128 == 0, K % 32 == 0, N edge handled by row-clamp (garbage columns never stored).
// 256 threads, tile 128x128, BK=32, double-buffered LDS with async global->LDS staging.
__global__ void k_gemm_bf16(const unsigned short* __restrict__ A,
                            const unsigned short* __restrict__ B,
                            float* __restrict__ C,
                            int M, int N, int K, int lda, int ldb, int ldc) {
    __shared__ unsigned short As[2][128 * 40];
    __shared__ unsigned short Bs[2][128 * 40];
    const int tid = threadIdx.x, lane = tid & 31, w = tid >> 5;
    const int wm = w >> 1, wn = w & 1;            // 4x2 wave grid, each wave 32x64
    const int blockN = blockIdx.x * 128, blockM = blockIdx.y * 128;

    // per-thread staging coordinates: 2 chunks of 8 bf16 per matrix
    int row0 = (tid * 2) >> 2, kc0 = ((tid * 2) & 3) * 8;
    int row1 = (tid * 2 + 1) >> 2, kc1 = ((tid * 2 + 1) & 3) * 8;
    const int arow0 = blockM + row0, arow1 = blockM + row1;
    int br0 = blockN + row0; if (br0 >= N) br0 = N - 1;   // clamp: junk cols never stored
    int br1 = blockN + row1; if (br1 >= N) br1 = N - 1;

    v8f acc[2][4];
    v8f zero = {};
#pragma unroll
    for (int mi = 0; mi < 2; ++mi)
#pragma unroll
        for (int ni = 0; ni < 4; ++ni) acc[mi][ni] = zero;

    auto stage = [&](int buf, int k0) {
#if HAVE_ASYNC_LDS
        __builtin_amdgcn_global_load_async_to_lds_b128(
            (g_v4i_p)(A + (size_t)arow0 * lda + k0 + kc0),
            (l_v4i_p)&As[buf][row0 * 40 + kc0], 0, 0);
        __builtin_amdgcn_global_load_async_to_lds_b128(
            (g_v4i_p)(A + (size_t)arow1 * lda + k0 + kc1),
            (l_v4i_p)&As[buf][row1 * 40 + kc1], 0, 0);
        __builtin_amdgcn_global_load_async_to_lds_b128(
            (g_v4i_p)(B + (size_t)br0 * ldb + k0 + kc0),
            (l_v4i_p)&Bs[buf][row0 * 40 + kc0], 0, 0);
        __builtin_amdgcn_global_load_async_to_lds_b128(
            (g_v4i_p)(B + (size_t)br1 * ldb + k0 + kc1),
            (l_v4i_p)&Bs[buf][row1 * 40 + kc1], 0, 0);
#else
        uint4 a0 = *(const uint4*)(A + (size_t)arow0 * lda + k0 + kc0);
        uint4 a1 = *(const uint4*)(A + (size_t)arow1 * lda + k0 + kc1);
        uint4 b0 = *(const uint4*)(B + (size_t)br0 * ldb + k0 + kc0);
        uint4 b1 = *(const uint4*)(B + (size_t)br1 * ldb + k0 + kc1);
        *(uint4*)(&As[buf][row0 * 40 + kc0]) = a0;
        *(uint4*)(&As[buf][row1 * 40 + kc1]) = a1;
        *(uint4*)(&Bs[buf][row0 * 40 + kc0]) = b0;
        *(uint4*)(&Bs[buf][row1 * 40 + kc1]) = b1;
#endif
    };

    const int nk = K >> 5;
    stage(0, 0);
#if HAVE_ASYNC_LDS
    __builtin_amdgcn_s_wait_asynccnt(0);
#endif
    __syncthreads();

    for (int it = 0; it < nk; ++it) {
        const int cur = it & 1;
        if (it + 1 < nk) stage(cur ^ 1, (it + 1) * 32);

        v16bf af[2], bf[4];
#pragma unroll
        for (int mi = 0; mi < 2; ++mi)
            af[mi] = load_frag_a(&As[cur][(wm * 32 + mi * 16) * 40], 40, lane);
#pragma unroll
        for (int ni = 0; ni < 4; ++ni)
            bf[ni] = load_frag_b(&Bs[cur][(wn * 64 + ni * 16) * 40], 40, lane);
#pragma unroll
        for (int mi = 0; mi < 2; ++mi)
#pragma unroll
            for (int ni = 0; ni < 4; ++ni)
                acc[mi][ni] = wmma_bf16(af[mi], bf[ni], acc[mi][ni]);

#if HAVE_ASYNC_LDS
        __builtin_amdgcn_s_wait_asynccnt(0);
#endif
        __syncthreads();
    }

    const int half = lane >> 4, nl = lane & 15;
#pragma unroll
    for (int mi = 0; mi < 2; ++mi)
#pragma unroll
        for (int ni = 0; ni < 4; ++ni) {
            int n = blockN + wn * 64 + ni * 16 + nl;
            if (n < N) {
#pragma unroll
                for (int i = 0; i < 8; ++i) {
                    int m = blockM + wm * 32 + mi * 16 + half * 8 + i;
                    C[(size_t)m * ldc + n] = acc[mi][ni][i];
                }
            }
        }
}

// ---------------- kernel 2: causal conv(4) + SiLU + split + softplus(dt) ----------------
__global__ void k_conv_split(const float* __restrict__ zx,
                             const float* __restrict__ conv_w,
                             const float* __restrict__ conv_b,
                             const float* __restrict__ dt_bias,
                             float* __restrict__ xf32,
                             unsigned short* __restrict__ xT,
                             unsigned short* __restrict__ Bbf,
                             unsigned short* __restrict__ BT,
                             unsigned short* __restrict__ Cbf,
                             float* __restrict__ dtarr) {
    const int l = blockIdx.x;
    const int c = blockIdx.y * 256 + threadIdx.x;
    if (c < CONV_DIM) {
        float acc = conv_b[c];
#pragma unroll
        for (int j = 0; j < 4; ++j) {
            int l0 = l - 3 + j;
            if (l0 >= 0) acc += conv_w[c * 4 + j] * zx[(size_t)l0 * D_IN_PROJ + D_INNER + c];
        }
        float y = silu(acc);
        if (c < D_INNER) {
            xf32[(size_t)l * D_INNER + c] = y;
            xT[(size_t)c * SEQLEN + l] = f2bf(y);       // c == h*64+p
        } else if (c < D_INNER + D_STATE) {
            int n = c - D_INNER;
            Bbf[(size_t)l * D_STATE + n] = f2bf(y);
            BT[(size_t)n * SEQLEN + l] = f2bf(y);
        } else {
            int n = c - D_INNER - D_STATE;
            Cbf[(size_t)l * D_STATE + n] = f2bf(y);
        }
    } else if (c < CONV_DIM + NHEADS) {
        int h = c - CONV_DIM;
        float v = zx[(size_t)l * D_IN_PROJ + D_INNER + CONV_DIM + h] + dt_bias[h];
        dtarr[(size_t)h * SEQLEN + l] = softplus_f(v);
    }
}

// ---------------- kernel 3: per-(chunk,head) inclusive scan of dA = dt*A ----------------
__global__ void k_scan(const float* __restrict__ dtarr,
                       const float* __restrict__ A_log,
                       float* __restrict__ Acum,
                       float* __restrict__ cdec) {
    const int c = blockIdx.x, h = blockIdx.y, tid = threadIdx.x;
    __shared__ float s[CHUNK];
    float Aneg = -expf(A_log[h]);
    s[tid] = dtarr[(size_t)h * SEQLEN + c * CHUNK + tid] * Aneg;
    __syncthreads();
    for (int off = 1; off < CHUNK; off <<= 1) {
        float t = (tid >= off) ? s[tid - off] : 0.0f;
        __syncthreads();
        s[tid] += t;
        __syncthreads();
    }
    Acum[(size_t)h * SEQLEN + c * CHUNK + tid] = s[tid];
    if (tid == CHUNK - 1) cdec[h * NCHUNK + c] = expf(s[tid]);
}

// ---------------- kernel 4: chunk states  states[c,h,p,n] = sum_l xT[p,l]*scale[l]*BT[n,l]
__global__ void k_states(const unsigned short* __restrict__ xT,
                         const unsigned short* __restrict__ BT,
                         const float* __restrict__ dtarr,
                         const float* __restrict__ Acum,
                         float* __restrict__ statesf) {
    const int c = blockIdx.x, h = blockIdx.y;
    const int tid = threadIdx.x, lane = tid & 31, w = tid >> 5;
    const int pb = w & 3, noff = (w >> 2) * 64;
    __shared__ float scale_s[CHUNK];
    __shared__ unsigned short BTs[128 * 136];

    {
        int l = tid, gl = c * CHUNK + l;
        float alast = Acum[(size_t)h * SEQLEN + c * CHUNK + (CHUNK - 1)];
        float a = Acum[(size_t)h * SEQLEN + gl];
        scale_s[l] = dtarr[(size_t)h * SEQLEN + gl] * expf(alast - a);
    }
    __syncthreads();

    v8f acc[4];
    v8f zero = {};
#pragma unroll
    for (int j = 0; j < 4; ++j) acc[j] = zero;

    for (int halfk = 0; halfk < 2; ++halfk) {
        // stage scaled B^T half-chunk: 128 n x 128 l
        for (int i = 0; i < 64; ++i) {
            int idx = tid + i * 256;
            int n = idx >> 7, l = idx & 127;
            int gl = c * CHUNK + halfk * 128 + l;
            float v = bf2f(BT[(size_t)n * SEQLEN + gl]) * scale_s[halfk * 128 + l];
            BTs[n * 136 + l] = f2bf(v);
        }
        __syncthreads();
#pragma unroll
        for (int kk = 0; kk < 128; kk += 32) {
            v16bf a = load_frag_a(xT + (size_t)(h * HEADDIM + pb * 16) * SEQLEN +
                                      c * CHUNK + halfk * 128 + kk, SEQLEN, lane);
            v16bf b[4];
#pragma unroll
            for (int j = 0; j < 4; ++j)
                b[j] = load_frag_b(&BTs[(noff + j * 16) * 136 + kk], 136, lane);
#pragma unroll
            for (int j = 0; j < 4; ++j) acc[j] = wmma_bf16(a, b[j], acc[j]);
        }
        __syncthreads();
    }

    const int half = lane >> 4, nl = lane & 15;
    float* base = statesf + ((size_t)(c * NHEADS + h)) * HEADDIM * D_STATE;
#pragma unroll
    for (int j = 0; j < 4; ++j)
#pragma unroll
        for (int i = 0; i < 8; ++i) {
            int p = pb * 16 + half * 8 + i;
            int n = noff + j * 16 + nl;
            base[(size_t)p * D_STATE + n] = acc[j][i];
        }
}

// ---------------- kernel 5: inter-chunk recurrence (per head) ----------------
__global__ void k_recur(const float* __restrict__ statesf,
                        const float* __restrict__ cdec,
                        unsigned short* __restrict__ prevbf) {
    const int h = blockIdx.x, tid = threadIdx.x;
    for (int pn = tid; pn < HEADDIM * D_STATE; pn += 256) {
        float run = 0.0f;
        for (int c = 0; c < NCHUNK; ++c) {
            size_t off = ((size_t)(c * NHEADS + h)) * HEADDIM * D_STATE + pn;
            prevbf[off] = f2bf(run);                 // state entering chunk c
            run = run * cdec[h * NCHUNK + c] + statesf[off];
        }
    }
}

// ---------------- kernel 6: per-chunk G = Cc * Bc^T (256x256, K=128) ----------------
__global__ void k_gmat(const unsigned short* __restrict__ Cbf,
                       const unsigned short* __restrict__ Bbf,
                       float* __restrict__ G) {
    const int c = blockIdx.z;
    const int blockM = blockIdx.y * 128, blockN = blockIdx.x * 128;  // M=l, N=s
    const int tid = threadIdx.x, lane = tid & 31, w = tid >> 5;
    const int wm = w >> 1, wn = w & 1;

    v8f acc[2][4];
    v8f zero = {};
#pragma unroll
    for (int mi = 0; mi < 2; ++mi)
#pragma unroll
        for (int ni = 0; ni < 4; ++ni) acc[mi][ni] = zero;

#pragma unroll
    for (int kk = 0; kk < D_STATE; kk += 32) {
        v16bf af[2], bf[4];
#pragma unroll
        for (int mi = 0; mi < 2; ++mi)
            af[mi] = load_frag_a(Cbf + (size_t)(c * CHUNK + blockM + wm * 32 + mi * 16) * D_STATE + kk,
                                 D_STATE, lane);
#pragma unroll
        for (int ni = 0; ni < 4; ++ni)
            bf[ni] = load_frag_b(Bbf + (size_t)(c * CHUNK + blockN + wn * 64 + ni * 16) * D_STATE + kk,
                                 D_STATE, lane);
#pragma unroll
        for (int mi = 0; mi < 2; ++mi)
#pragma unroll
            for (int ni = 0; ni < 4; ++ni)
                acc[mi][ni] = wmma_bf16(af[mi], bf[ni], acc[mi][ni]);
    }

    const int half = lane >> 4, nl = lane & 15;
#pragma unroll
    for (int mi = 0; mi < 2; ++mi)
#pragma unroll
        for (int ni = 0; ni < 4; ++ni)
#pragma unroll
            for (int i = 0; i < 8; ++i) {
                int l = blockM + wm * 32 + mi * 16 + half * 8 + i;
                int s = blockN + wn * 64 + ni * 16 + nl;
                G[(size_t)c * CHUNK * CHUNK + (size_t)l * CHUNK + s] = acc[mi][ni][i];
            }
}

// ---------------- kernel 7: Y = exp(Acum)*(Cc*prev^T) + M*x + D*x  (per chunk,head) ----
__global__ void k_ychunk(const unsigned short* __restrict__ Cbf,
                         const unsigned short* __restrict__ prevbf,
                         const unsigned short* __restrict__ xT,
                         const float* __restrict__ G,
                         const float* __restrict__ Acum,
                         const float* __restrict__ dtarr,
                         const float* __restrict__ Dvec,
                         const float* __restrict__ xf32,
                         float* __restrict__ Yf) {
    const int c = blockIdx.x, h = blockIdx.y;
    const int tid = threadIdx.x, lane = tid & 31, w = tid >> 5;
    __shared__ float eA[CHUNK], dts[CHUNK], expA[CHUNK];
    __shared__ unsigned short Ms[256 * 72];

    {
        int l = tid, gl = c * CHUNK + l;
        float a = Acum[(size_t)h * SEQLEN + gl];
        eA[l] = a;
        expA[l] = expf(a);
        dts[l] = dtarr[(size_t)h * SEQLEN + gl];
    }
    __syncthreads();

    v8f acc[2][4];
    v8f zero = {};
#pragma unroll
    for (int mi = 0; mi < 2; ++mi)
#pragma unroll
        for (int ni = 0; ni < 4; ++ni) acc[mi][ni] = zero;

    // ---- phase 1: off-diagonal  Cc (256x128) * prev^T (128x64)
    const unsigned short* prevCH = prevbf + ((size_t)(c * NHEADS + h)) * HEADDIM * D_STATE;
#pragma unroll
    for (int kk = 0; kk < D_STATE; kk += 32) {
        v16bf af[2], bf[4];
#pragma unroll
        for (int mi = 0; mi < 2; ++mi)
            af[mi] = load_frag_a(Cbf + (size_t)(c * CHUNK + w * 32 + mi * 16) * D_STATE + kk,
                                 D_STATE, lane);
#pragma unroll
        for (int ni = 0; ni < 4; ++ni)
            bf[ni] = load_frag_b(prevCH + (size_t)(ni * 16) * D_STATE + kk, D_STATE, lane);
#pragma unroll
        for (int mi = 0; mi < 2; ++mi)
#pragma unroll
            for (int ni = 0; ni < 4; ++ni)
                acc[mi][ni] = wmma_bf16(af[mi], bf[ni], acc[mi][ni]);
    }
    // row-scale by exp(Acum[l])
    {
        const int half = lane >> 4;
#pragma unroll
        for (int mi = 0; mi < 2; ++mi) {
#pragma unroll
            for (int i = 0; i < 8; ++i) {
                float s = expA[w * 32 + mi * 16 + half * 8 + i];
#pragma unroll
                for (int ni = 0; ni < 4; ++ni) acc[mi][ni][i] *= s;
            }
        }
    }

    // ---- phase 2: diagonal  M (256x256, masked/decayed) * x (256x64), s-tiles of 64
    for (int st = 0; st < 4; ++st) {
        int s0 = st * 64;
        for (int i = 0; i < 64; ++i) {
            int idx = tid + i * 256;
            int l = idx >> 6, sj = idx & 63, s = s0 + sj;
            float m = 0.0f;
            if (s <= l) {
                float g = G[(size_t)c * CHUNK * CHUNK + (size_t)l * CHUNK + s];
                m = g * expf(eA[l] - eA[s]) * dts[s];
            }
            Ms[l * 72 + sj] = f2bf(m);
        }
        __syncthreads();
#pragma unroll
        for (int kk = 0; kk < 64; kk += 32) {
            v16bf af[2], bf[4];
#pragma unroll
            for (int mi = 0; mi < 2; ++mi)
                af[mi] = load_frag_a(&Ms[(w * 32 + mi * 16) * 72 + kk], 72, lane);
#pragma unroll
            for (int ni = 0; ni < 4; ++ni)
                bf[ni] = load_frag_b(xT + (size_t)(h * HEADDIM + ni * 16) * SEQLEN +
                                         c * CHUNK + s0 + kk, SEQLEN, lane);
#pragma unroll
            for (int mi = 0; mi < 2; ++mi)
#pragma unroll
                for (int ni = 0; ni < 4; ++ni)
                    acc[mi][ni] = wmma_bf16(af[mi], bf[ni], acc[mi][ni]);
        }
        __syncthreads();
    }

    // ---- epilogue: + D[h]*x, store Y (l, h*64+p)
    const float Dh = Dvec[h];
    const int half = lane >> 4, nl = lane & 15;
#pragma unroll
    for (int mi = 0; mi < 2; ++mi)
#pragma unroll
        for (int ni = 0; ni < 4; ++ni)
#pragma unroll
            for (int i = 0; i < 8; ++i) {
                int l = w * 32 + mi * 16 + half * 8 + i;
                int p = ni * 16 + nl;
                size_t gl = (size_t)(c * CHUNK + l);
                size_t e = (size_t)h * HEADDIM + p;
                float y = acc[mi][ni][i] + Dh * xf32[gl * D_INNER + e];
                Yf[gl * D_INNER + e] = y;
            }
}

// ---------------- kernel 8: gated RMSNorm ----------------
__global__ void k_norm(const float* __restrict__ Yf,
                       const float* __restrict__ zx,
                       const float* __restrict__ norm_w,
                       unsigned short* __restrict__ ynorm) {
    const int l = blockIdx.x, tid = threadIdx.x;
    __shared__ float red[256];
    float g[16];
    float sum = 0.0f;
#pragma unroll
    for (int i = 0; i < 16; ++i) {
        int e = tid + i * 256;
        float z = zx[(size_t)l * D_IN_PROJ + e];
        float v = Yf[(size_t)l * D_INNER + e] * silu(z);
        g[i] = v;
        sum += v * v;
    }
    red[tid] = sum;
    __syncthreads();
    for (int off = 128; off > 0; off >>= 1) {
        if (tid < off) red[tid] += red[tid + off];
        __syncthreads();
    }
    float rs = rsqrtf(red[0] / (float)D_INNER + EPS);
#pragma unroll
    for (int i = 0; i < 16; ++i) {
        int e = tid + i * 256;
        ynorm[(size_t)l * D_INNER + e] = f2bf(g[i] * rs * norm_w[e]);
    }
}

// ---------------- launch ----------------
static inline size_t align_up(size_t x) { return (x + 255) & ~(size_t)255; }

extern "C" void kernel_launch(void* const* d_in, const int* in_sizes, int n_in,
                              void* d_out, int out_size, void* d_ws, size_t ws_size,
                              hipStream_t stream) {
    const float* u       = (const float*)d_in[0];
    const float* W_in    = (const float*)d_in[1];
    const float* conv_w  = (const float*)d_in[2];
    const float* conv_b  = (const float*)d_in[3];
    const float* dt_bias = (const float*)d_in[4];
    const float* A_log   = (const float*)d_in[5];
    const float* Dvec    = (const float*)d_in[6];
    const float* norm_w  = (const float*)d_in[7];
    const float* W_out   = (const float*)d_in[8];
    float* out = (float*)d_out;

    char* p = (char*)d_ws;
    size_t off = 0;
    auto take = [&](size_t bytes) { size_t o = off; off = align_up(off + bytes); return (void*)(p + o); };

    unsigned short* ubf    = (unsigned short*)take((size_t)SEQLEN * D_MODEL * 2);
    unsigned short* winbf  = (unsigned short*)take((size_t)D_IN_PROJ * D_MODEL * 2);
    unsigned short* woutbf = (unsigned short*)take((size_t)D_MODEL * D_INNER * 2);
    float*          zx     = (float*)take((size_t)SEQLEN * D_IN_PROJ * 4);
    float*          xf32   = (float*)take((size_t)SEQLEN * D_INNER * 4);
    unsigned short* xT     = (unsigned short*)take((size_t)D_INNER * SEQLEN * 2);
    unsigned short* Bbf    = (unsigned short*)take((size_t)SEQLEN * D_STATE * 2);
    unsigned short* BT     = (unsigned short*)take((size_t)D_STATE * SEQLEN * 2);
    unsigned short* Cbf    = (unsigned short*)take((size_t)SEQLEN * D_STATE * 2);
    float*          dtarr  = (float*)take((size_t)NHEADS * SEQLEN * 4);
    float*          Acum   = (float*)take((size_t)NHEADS * SEQLEN * 4);
    float*          cdec   = (float*)take((size_t)NHEADS * NCHUNK * 4);
    float*          states = (float*)take((size_t)NCHUNK * NHEADS * HEADDIM * D_STATE * 4);
    unsigned short* prevbf = (unsigned short*)take((size_t)NCHUNK * NHEADS * HEADDIM * D_STATE * 2);
    float*          G      = (float*)take((size_t)NCHUNK * CHUNK * CHUNK * 4);
    float*          Yf     = (float*)take((size_t)SEQLEN * D_INNER * 4);
    unsigned short* ynorm  = (unsigned short*)take((size_t)SEQLEN * D_INNER * 2);
    (void)ws_size; (void)n_in; (void)in_sizes; (void)out_size;

    // 1) casts to bf16
    {
        int n;
        n = SEQLEN * D_MODEL;
        k_cast_bf16<<<(n + 255) / 256, 256, 0, stream>>>(u, ubf, n);
        n = D_IN_PROJ * D_MODEL;
        k_cast_bf16<<<(n + 255) / 256, 256, 0, stream>>>(W_in, winbf, n);
        n = D_MODEL * D_INNER;
        k_cast_bf16<<<(n + 255) / 256, 256, 0, stream>>>(W_out, woutbf, n);
    }
    // 2) in_proj: zx = u * W_in^T  (M=4096, N=8512, K=2048)
    k_gemm_bf16<<<dim3((D_IN_PROJ + 127) / 128, SEQLEN / 128), 256, 0, stream>>>(
        ubf, winbf, zx, SEQLEN, D_IN_PROJ, D_MODEL, D_MODEL, D_MODEL, D_IN_PROJ);
    // 3) conv + silu + split + dt
    k_conv_split<<<dim3(SEQLEN, (CONV_DIM + NHEADS + 255) / 256), 256, 0, stream>>>(
        zx, conv_w, conv_b, dt_bias, xf32, xT, Bbf, BT, Cbf, dtarr);
    // 4) cumulative dA
    k_scan<<<dim3(NCHUNK, NHEADS), CHUNK, 0, stream>>>(dtarr, A_log, Acum, cdec);
    // 5) chunk states
    k_states<<<dim3(NCHUNK, NHEADS), 256, 0, stream>>>(xT, BT, dtarr, Acum, states);
    // 6) inter-chunk recurrence
    k_recur<<<NHEADS, 256, 0, stream>>>(states, cdec, prevbf);
    // 7) G = C * B^T per chunk
    k_gmat<<<dim3(2, 2, NCHUNK), 256, 0, stream>>>(Cbf, Bbf, G);
    // 8) Y chunks (diag + off-diag + D*x)
    k_ychunk<<<dim3(NCHUNK, NHEADS), 256, 0, stream>>>(Cbf, prevbf, xT, G, Acum, dtarr,
                                                      Dvec, xf32, Yf);
    // 9) gated RMSNorm
    k_norm<<<SEQLEN, 256, 0, stream>>>(Yf, zx, norm_w, ynorm);
    // 10) out_proj: out = ynorm * W_out^T  (M=4096, N=2048, K=4096)
    k_gemm_bf16<<<dim3(D_MODEL / 128, SEQLEN / 128), 256, 0, stream>>>(
        ynorm, woutbf, out, SEQLEN, D_MODEL, D_INNER, D_INNER, D_INNER, D_MODEL);
}
